// Network_82386062672409
// MI455X (gfx1250) — matrix operand order repaired and checked by reading, not compiled, not run on hardware
//
#include <hip/hip_runtime.h>
#include <hip/hip_bf16.h>
#include <math.h>

typedef __attribute__((ext_vector_type(16))) _Float16 v16h;
typedef __attribute__((ext_vector_type(8)))  _Float16 v8h;
typedef __attribute__((ext_vector_type(8)))  float    v8f;

#define SWISH_C 1.6768589f
#define SIG_C   1.8436479f
#define TANH_C  1.5926549f
#define PI_F    3.14159265358979f

// ---------------------------------------------------------------- utilities
__global__ void zero_u32_kernel(unsigned* p, long n) {
    long i = (long)blockIdx.x * blockDim.x + threadIdx.x;
    if (i < n) p[i] = 0u;
}

// Repack B operand for WMMA: in (K x Nq, f32 row-major) -> out (Nq x Kpad, f16),
// out[q*Kpad + k] = (k < K) ? in[k*Nq + q] : 0
__global__ void repackT_kernel(const float* in, _Float16* out, int K, int Kpad, long Nq)
{
    long idx = (long)blockIdx.x * blockDim.x + threadIdx.x;
    if (idx >= Nq * Kpad) return;
    int  k = (int)(idx % Kpad);
    long q = idx / Kpad;
    float v = (k < K) ? in[(long)k * Nq + q] : 0.f;
    out[idx] = (_Float16)v;
}

// ---------------------------------------------------------------- edge geometry
__global__ void edge_geom_kernel(const float* pos, const int* z,
                                 const int* esrc, const int* edst,
                                 float* esh, _Float16* emb16, int* vofe, int* cnt, int E)
{
    int e = blockIdx.x * blockDim.x + threadIdx.x;
    if (e >= E) return;
    int s = esrc[e], d = edst[e];
    float vx = pos[s*3+0] - pos[d*3+0];
    float vy = pos[s*3+1] - pos[d*3+1];
    float vz = pos[s*3+2] - pos[d*3+2];
    float r = sqrtf(vx*vx + vy*vy + vz*vz + 1e-24f);
    float inv = 1.f / r;
    float c = (cosf(PI_F * r / 10.f) + 1.f) * 0.5f;
    float f = c * 0.2236067977f;           // cutoff / sqrt(NNEIGH=20)
    const float SQ3 = 1.7320508075f;
    esh[e*4+0] = f;
    esh[e*4+1] = f * SQ3 * vx * inv;
    esh[e*4+2] = f * SQ3 * vy * inv;
    esh[e*4+3] = f * SQ3 * vz * inv;
    float step = 10.f / 49.f;
    for (int j = 0; j < 64; ++j) {
        float val = 0.f;
        if (j < 50) { float t = (r - j*step)/step; val = expf(-0.5f*t*t); }
        emb16[(long)e*64 + j] = (_Float16)val;
    }
    int tab[10] = {-1,0,-1,-1,-1,-1,1,2,3,4};
    int v = 5*tab[z[s]] + tab[z[d]];
    vofe[e] = v;
    atomicAdd(&cnt[v], 1);
}

__global__ void hinit_scatter_kernel(const int* esrc, const float* esh, float* hA, int E)
{
    int e = blockIdx.x * blockDim.x + threadIdx.x;
    if (e >= E) return;
    int s = esrc[e];
    for (int k = 0; k < 4; ++k) atomicAdd(&hA[(long)s*4 + k], esh[e*4+k]);
}

__global__ void hinit_set_kernel(float* hA, int N)
{
    int n = blockIdx.x * blockDim.x + threadIdx.x;
    if (n < N) hA[(long)n*4 + 0] = 1.f;
}

__global__ void prefix25_kernel(const int* cnt, int* offs)
{
    if (threadIdx.x == 0 && blockIdx.x == 0) {
        int a = 0;
        for (int i = 0; i < 25; ++i) { offs[i] = a; a += cnt[i]; }
        offs[25] = a;
    }
}

__global__ void bucket_fill_kernel(const int* vofe, const int* offs, int* cur, int* elist, int E)
{
    int e = blockIdx.x * blockDim.x + threadIdx.x;
    if (e >= E) return;
    int v = vofe[e];
    int p = atomicAdd(&cur[v], 1);
    elist[offs[v] + p] = e;
}

// ---------------------------------------------------------------- fragment helpers
__device__ inline v16h load_a_frag(const _Float16* Ar, int k0, int hi)
{
    v8h a0 = *(const v8h*)(Ar + k0 + 16*hi);
    v8h a1 = *(const v8h*)(Ar + k0 + 16*hi + 8);
    return __builtin_shufflevector(a0, a1, 0,1,2,3,4,5,6,7,8,9,10,11,12,13,14,15);
}

// ---------------------------------------------------------------- WMMA GEMM + SiLU
// O (f16, M x N) = SWISH_C * silu( scale * A(f16, M x K) @ B ),  BT: (N x K) f16 transposed
template<int KSTEPS>
__global__ void gemm_silu_kernel(const _Float16* A, int lda, const _Float16* BT, int ldbt,
                                 _Float16* O, int ldo, float scale, int Mvalid)
{
    int mt = blockIdx.x, nt = blockIdx.y;
    int lane = threadIdx.x;
    int lo = lane & 15, hi = lane >> 4;
    int arow = mt*16 + lo;
    int bcol = nt*16 + lo;
    v8f acc = {};
    const _Float16* Ar = A  + (long)arow * lda;
    const _Float16* Br = BT + (long)bcol * ldbt;
#pragma unroll
    for (int s = 0; s < KSTEPS; ++s) {
        int k0 = s*32;
        v16h av = load_a_frag(Ar, k0, hi);
        v16h bv = *(const v16h*)(Br + k0 + 16*hi);
        acc = __builtin_amdgcn_wmma_f32_16x16x32_f16(false, av, false, bv, (short)0, acc, false, false);
    }
#pragma unroll
    for (int g = 0; g < 8; ++g) {
        int row = mt*16 + g + 8*hi;
        if (row < Mvalid) {
            float v = acc[g] * scale;
            float y = SWISH_C * v / (1.f + expf(-v));
            O[(long)row*ldo + bcol] = (_Float16)y;
        }
    }
}

// ---------------------------------------------------------------- tensor product kernel
struct TpPath { int off; int m1; int type; int i1b; int ob; float fnorm; };
struct TpDesc { int npaths; int ncols; int xld; int aggld; TpPath p[10]; };

// tpwT: (numel x 128) f16 : tpwT[q*128 + k] = tp_weight[k, q]
__global__ void tp_kernel(const _Float16* wq16, const _Float16* tpwT, TpDesc D,
                          const float* xbuf, float* agg, const float* esh,
                          const int* elist, const int* offs, const int* cnt,
                          const int* esrc, const int* edst)
{
    int v = blockIdx.z;
    int count = cnt[v];
    int rb = blockIdx.y * 16;
    if (rb >= count) return;                       // uniform exit (covers count==0)
    int cb = blockIdx.x * 16;
    int lane = threadIdx.x;
    int lo = lane & 15, hi = lane >> 4;
    int base = offs[v];

    int rA = rb + lo; if (rA >= count) rA = count - 1;
    int eA = elist[base + rA];

    int c = cb + lo;
    bool cvalid = (c < D.ncols);
    int cc = cvalid ? c : (D.ncols - 1);
    int p = 0, u = cc;
    while (p < D.npaths - 1 && u >= D.p[p].m1) { u -= D.p[p].m1; ++p; }
    long q = (long)D.p[p].off + (long)u*25 + v;    // gathered tp_weight column
    const _Float16* Ar = wq16 + (long)eA * 128;
    const _Float16* Br = tpwT + q * 128;
    __builtin_prefetch(Br);

    v8f acc = {};
#pragma unroll
    for (int s = 0; s < 4; ++s) {
        int k0 = s*32;
        v16h av = load_a_frag(Ar, k0, hi);
        v16h bv = *(const v16h*)(Br + k0 + 16*hi);
        acc = __builtin_amdgcn_wmma_f32_16x16x32_f16(false, av, false, bv, (short)0, acc, false, false);
    }
    if (!cvalid) return;
    TpPath P = D.p[p];
#pragma unroll
    for (int g = 0; g < 8; ++g) {
        int r = rb + g + 8*hi;
        if (r >= count) continue;
        int e = elist[base + r];
        float w = acc[g];
        float e0 = esh[e*4+0], e1 = esh[e*4+1], e2 = esh[e*4+2], e3 = esh[e*4+3];
        int ns = esrc[e], nd = edst[e];
        const float* xr = xbuf + (long)ns * D.xld;
        float* ar = agg + (long)nd * D.aggld;
        float f = P.fnorm * w;
        if (P.type == 0) {                 // (0,0)->0 : scalar*scalar
            atomicAdd(&ar[P.ob + u], f * xr[P.i1b + u] * e0);
        } else if (P.type == 1) {          // (0,1)->1
            float t = f * xr[P.i1b + u];
            atomicAdd(&ar[P.ob + u*3+0], t*e1);
            atomicAdd(&ar[P.ob + u*3+1], t*e2);
            atomicAdd(&ar[P.ob + u*3+2], t*e3);
        } else if (P.type == 2) {          // (1,0)->1
            float t = f * e0;
            atomicAdd(&ar[P.ob + u*3+0], t*xr[P.i1b + u*3+0]);
            atomicAdd(&ar[P.ob + u*3+1], t*xr[P.i1b + u*3+1]);
            atomicAdd(&ar[P.ob + u*3+2], t*xr[P.i1b + u*3+2]);
        } else if (P.type == 3) {          // (1,1)->0 : dot
            float bx = xr[P.i1b+u*3+0], by = xr[P.i1b+u*3+1], bz = xr[P.i1b+u*3+2];
            atomicAdd(&ar[P.ob + u], f*(bx*e1 + by*e2 + bz*e3));
        } else {                           // (1,1)->1 : cross
            float bx = xr[P.i1b+u*3+0], by = xr[P.i1b+u*3+1], bz = xr[P.i1b+u*3+2];
            atomicAdd(&ar[P.ob + u*3+0], f*(by*e3 - bz*e2));
            atomicAdd(&ar[P.ob + u*3+1], f*(bz*e1 - bx*e3));
            atomicAdd(&ar[P.ob + u*3+2], f*(bx*e2 - by*e1));
        }
    }
}

// ---------------------------------------------------------------- node linears
// y[n, w*nc+i] (+)= (1/sqrt(mi)) * sum_u X[n, segbase + u*nc + i] * W[u_global, w]
__global__ void linear_kernel(const float* X, int xld, float* Y, int yld, int ybase,
                              const float* W, int mo, int nc,
                              int s0b, int s0m, int s1b, int s1m,
                              float accumF, int N)
{
    int idx = blockIdx.x * blockDim.x + threadIdx.x;
    int total = N * mo * nc;
    if (idx >= total) return;
    int i = idx % nc;
    int w = (idx / nc) % mo;
    int n = idx / (nc * mo);
    const float* xr = X + (long)n * xld;
    float acc = 0.f;
    int ug = 0;
    for (int u = 0; u < s0m; ++u, ++ug) acc += xr[s0b + u*nc + i] * W[(long)ug*mo + w];
    for (int u = 0; u < s1m; ++u, ++ug) acc += xr[s1b + u*nc + i] * W[(long)ug*mo + w];
    float val = acc * rsqrtf((float)(s0m + s1m));
    float* y = &Y[(long)n*yld + ybase + w*nc + i];
    if (accumF != 0.f) *y += val * accumF; else *y = val;
}

// ---------------------------------------------------------------- gate
__global__ void gate_kernel(const float* Hin, int inld, float* Hout, int outld,
                            int nsilu, int ntanh, int N)
{
    int Dd = nsilu + ntanh + 96;
    int idx = blockIdx.x * blockDim.x + threadIdx.x;
    if (idx >= N * Dd) return;
    int n = idx / Dd, c = idx % Dd;
    const float* hr = Hin + (long)n * inld;
    float* orow = Hout + (long)n * outld;
    if (c < nsilu) {
        float t = hr[c];
        orow[c] = SWISH_C * t / (1.f + expf(-t));
    } else if (c < nsilu + ntanh) {
        orow[c] = TANH_C * tanhf(hr[c]);
    } else {
        int j = c - nsilu - ntanh;           // 0..95
        int gb = nsilu + ntanh;
        float g = SIG_C / (1.f + expf(-hr[gb + (j/3)]));
        orow[c] = hr[gb + 32 + j] * g;
    }
}

// ---------------------------------------------------------------- readout
__global__ void readout_kernel(const float* hB, const int* batch, float* out, int N)
{
    int n = blockIdx.x * blockDim.x + threadIdx.x;
    if (n >= N) return;
    float s = hB[(long)n*2+0] + 0.5f * hB[(long)n*2+1] * hB[(long)n*2+1];
    atomicAdd(&out[batch[n]], s * 0.2236067977f);   // 1/sqrt(NATOMS=20)
}

// ================================================================ host
static TpPath mkpath(int off, int m1, int type, int i1b, int ob, int lo, float fan)
{
    float cs = (type == 0) ? 1.f : (type == 4 ? 0.4082482905f : 0.5773502692f);
    TpPath P; P.off = off; P.m1 = m1; P.type = type; P.i1b = i1b; P.ob = ob;
    P.fnorm = 5.f * sqrtf((2.f*lo + 1.f) / fan) * cs;
    return P;
}

extern "C" void kernel_launch(void* const* d_in, const int* in_sizes, int n_in,
                              void* d_out, int out_size, void* d_ws, size_t ws_size,
                              hipStream_t stream)
{
    const float* pos = (const float*)d_in[0];
    const float* P[47];
    for (int i = 0; i < 47; ++i) P[i] = (const float*)d_in[1 + i];
    const int* z     = (const int*)d_in[n_in-5];
    const int* batch = (const int*)d_in[n_in-4];
    const int* esrc  = (const int*)d_in[n_in-3];
    const int* edst  = (const int*)d_in[n_in-2];

    const int N  = in_sizes[0] / 3;          // 216
    const int E  = in_sizes[n_in-3];         // 3672
    const int EP = ((E + 15) / 16) * 16;     // 3680
    const int ET = EP / 16;                  // edge tile count

    // ---------------- workspace layout
    size_t o = 0;
    auto alloc = [&](size_t bytes) { size_t r = o; o += (bytes + 255) & ~(size_t)255; return r; };
    char* ws = (char*)d_ws;
    _Float16* emb16 = (_Float16*)(ws + alloc((size_t)EP*64*2));
    _Float16* h1_16 = (_Float16*)(ws + alloc((size_t)EP*128*2));
    _Float16* wq16  = (_Float16*)(ws + alloc((size_t)EP*128*2));
    _Float16* w1T   = (_Float16*)(ws + alloc((size_t)128*64*2));
    _Float16* w2T   = (_Float16*)(ws + alloc((size_t)128*128*2));
    _Float16* tpwT  = (_Float16*)(ws + alloc((size_t)28000*128*2));  // reused per layer
    float* esh  = (float*)(ws + alloc((size_t)E*4*4));
    int*   vofe = (int*)  (ws + alloc((size_t)E*4));
    int*   cnt  = (int*)  (ws + alloc(32*4));
    int*   offs = (int*)  (ws + alloc(32*4));
    int*   cur  = (int*)  (ws + alloc(32*4));
    int*   elist= (int*)  (ws + alloc((size_t)E*4));
    float* hA   = (float*)(ws + alloc((size_t)N*640*4));
    float* hB   = (float*)(ws + alloc((size_t)N*640*4));
    float* xbuf = (float*)(ws + alloc((size_t)N*640*4));
    float* agg  = (float*)(ws + alloc((size_t)N*2176*4));
    (void)ws_size;

    auto zero = [&](void* p, size_t bytes) {
        long n4 = (long)(bytes / 4);
        zero_u32_kernel<<<(int)((n4 + 255)/256), 256, 0, stream>>>((unsigned*)p, n4);
    };
    auto repack = [&](const float* in, _Float16* out, int K, int Kpad, long Nq) {
        long total = Nq * Kpad;
        repackT_kernel<<<(int)((total + 255)/256), 256, 0, stream>>>(in, out, K, Kpad, Nq);
    };
    auto lin = [&](const float* X, int xld, float* Y, int yld, int ybase,
                   const float* W, int mo, int nc,
                   int s0b, int s0m, int s1b, int s1m, float accumF) {
        int total = N * mo * nc;
        linear_kernel<<<(total + 255)/256, 256, 0, stream>>>(X, xld, Y, yld, ybase, W, mo, nc,
                                                             s0b, s0m, s1b, s1m, accumF, N);
    };

    // ---------------- preamble
    zero(emb16, (size_t)EP*64*2);
    zero(cnt, 32*4); zero(cur, 32*4);
    zero(hA, (size_t)N*640*4);
    zero(d_out, (size_t)out_size*4);

    int eb = (E + 255)/256;
    edge_geom_kernel<<<eb, 256, 0, stream>>>(pos, z, esrc, edst, esh, emb16, vofe, cnt, E);
    hinit_scatter_kernel<<<eb, 256, 0, stream>>>(esrc, esh, hA, E);
    hinit_set_kernel<<<1, 256, 0, stream>>>(hA, N);
    prefix25_kernel<<<1, 32, 0, stream>>>(cnt, offs);
    bucket_fill_kernel<<<eb, 256, 0, stream>>>(vofe, offs, cur, elist, E);

    // ---------------- radial MLP (WMMA, repacked f16 B)
    repack(P[45], w1T, 50, 64, 128);
    repack(P[46], w2T, 128, 128, 128);
    gemm_silu_kernel<2><<<dim3(ET, 8), 32, 0, stream>>>(emb16, 64,  w1T, 64,  h1_16, 128, 1.f, EP);
    gemm_silu_kernel<4><<<dim3(ET, 8), 32, 0, stream>>>(h1_16, 128, w2T, 128, wq16,  128, 0.0883883476f, EP);

    // ---------------- TP descriptors
    TpDesc D0; D0.npaths=5; D0.ncols=5;    D0.xld=4;   D0.aggld=7;
    D0.p[0]=mkpath(0,1,0,0,0,0,50);  D0.p[1]=mkpath(25,1,1,0,1,1,50);
    D0.p[2]=mkpath(50,1,2,1,1,1,50); D0.p[3]=mkpath(75,1,3,1,0,0,50);
    D0.p[4]=mkpath(100,1,4,1,4,1,25);
    for (int i=5;i<10;++i) D0.p[i]=D0.p[4];

    TpDesc D1; D1.npaths=8; D1.ncols=608;  D1.xld=352; D1.aggld=1152;
    D1.p[0]=mkpath(0,256,0,0,0,0,25);      D1.p[1]=mkpath(6400,256,1,0,256,1,25);
    D1.p[2]=mkpath(12800,16,2,256,1024,1,50); D1.p[3]=mkpath(13200,16,3,256,1072,0,25);
    D1.p[4]=mkpath(13600,16,4,256,1088,1,50); D1.p[5]=mkpath(14000,16,2,304,1088,1,50);
    D1.p[6]=mkpath(14400,16,3,304,1136,0,25); D1.p[7]=mkpath(14800,16,4,304,1024,1,50);
    for (int i=8;i<10;++i) D1.p[i]=D1.p[7];

    TpDesc D2; D2.npaths=10; D2.ncols=1120; D2.xld=608; D2.aggld=2176;
    D2.p[0]=mkpath(0,256,0,0,0,0,25);        D2.p[1]=mkpath(6400,256,1,0,256,1,25);
    D2.p[2]=mkpath(12800,256,0,256,1024,0,25); D2.p[3]=mkpath(19200,256,1,256,1280,1,25);
    D2.p[4]=mkpath(25600,16,2,512,2048,1,50);  D2.p[5]=mkpath(26000,16,3,512,2096,0,25);
    D2.p[6]=mkpath(26400,16,4,512,2112,1,50);  D2.p[7]=mkpath(26800,16,2,560,2112,1,50);
    D2.p[8]=mkpath(27200,16,3,560,2160,0,25);  D2.p[9]=mkpath(27600,16,4,560,2048,1,50);

    TpDesc D3; D3.npaths=4; D3.ncols=544;  D3.xld=608; D3.aggld=544;
    D3.p[0]=mkpath(0,256,0,0,0,0,25);     D3.p[1]=mkpath(6400,256,0,256,256,0,25);
    D3.p[2]=mkpath(12800,16,3,512,512,0,25); D3.p[3]=mkpath(13200,16,3,560,528,0,25);
    for (int i=4;i<10;++i) D3.p[i]=D3.p[3];

    // ================= layer 0 (h: N x 4) =================
    zero(hB, (size_t)N*384*4);
    lin(hA,4, hB,384,0,   P[5], 288,1, 0,1, 0,0, 0.f);     // si/0
    lin(hA,4, hB,384,288, P[6], 16,3,  1,1, 0,0, 0.f);     // si/1
    lin(hA,4, xbuf,4,0,   P[0], 1,1,   0,1, 0,0, 0.f);     // lin1/0
    lin(hA,4, xbuf,4,1,   P[1], 1,3,   1,1, 0,0, 0.f);     // lin1/1
    zero(agg, (size_t)N*7*4);
    repack(P[7], tpwT, 128, 128, 125);
    tp_kernel<<<dim3(1, ET, 25), 32, 0, stream>>>(wq16, tpwT, D0, xbuf, agg, esh, elist, offs, cnt, esrc, edst);
    lin(agg,7, hB,384,0,   P[2], 288,1, 0,1, 0,0, 0.1f);   // lin2/0
    lin(agg,7, hB,384,288, P[3], 16,3,  1,1, 0,0, 0.1f);   // lin2/1
    lin(agg,7, hB,384,336, P[4], 16,3,  4,1, 0,0, 0.1f);   // lin2/2
    gate_kernel<<<(N*352 + 255)/256, 256, 0, stream>>>(hB, 384, hA, 352, 256, 0, N);

    // ================= layer 1 (h: N x 352) =================
    zero(hB, (size_t)N*640*4);
    lin(hA,352, hB,640,0,   P[16], 256,1, 0,256, 0,0, 0.f);
    lin(hA,352, hB,640,512, P[17], 32,1,  0,256, 0,0, 0.f);
    lin(hA,352, hB,640,544, P[18], 16,3,  256,16, 0,0, 0.f);
    lin(hA,352, hB,640,592, P[19], 16,3,  304,16, 0,0, 0.f);
    lin(hA,352, xbuf,352,0,   P[8],  256,1, 0,256, 0,0, 0.f);
    lin(hA,352, xbuf,352,256, P[9],  16,3,  256,16, 0,0, 0.f);
    lin(hA,352, xbuf,352,304, P[10], 16,3,  304,16, 0,0, 0.f);
    zero(agg, (size_t)N*1152*4);
    repack(P[20], tpwT, 128, 128, 15200);
    tp_kernel<<<dim3(38, ET, 25), 32, 0, stream>>>(wq16, tpwT, D1, xbuf, agg, esh, elist, offs, cnt, esrc, edst);
    lin(agg,1152, hB,640,0,   P[11], 256,1, 0,256, 1072,16, 0.1f);
    lin(agg,1152, hB,640,256, P[12], 256,1, 1136,16, 0,0,   0.1f);
    lin(agg,1152, hB,640,512, P[13], 32,1,  0,256, 1072,16, 0.1f);
    lin(agg,1152, hB,640,544, P[14], 16,3,  256,256, 1024,16, 0.1f);
    lin(agg,1152, hB,640,592, P[15], 16,3,  1088,16, 0,0,   0.1f);
    gate_kernel<<<(N*608 + 255)/256, 256, 0, stream>>>(hB, 640, hA, 608, 256, 256, N);

    // ================= layer 2 (h: N x 608) =================
    zero(hB, (size_t)N*640*4);
    lin(hA,608, hB,640,0,   P[30], 256,1, 0,256,   0,0, 0.f);
    lin(hA,608, hB,640,256, P[31], 256,1, 256,256, 0,0, 0.f);
    lin(hA,608, hB,640,512, P[32], 32,1,  0,256,   0,0, 0.f);
    lin(hA,608, hB,640,544, P[33], 16,3,  512,16,  0,0, 0.f);
    lin(hA,608, hB,640,592, P[34], 16,3,  560,16,  0,0, 0.f);
    lin(hA,608, xbuf,608,0,   P[21], 256,1, 0,256,   0,0, 0.f);
    lin(hA,608, xbuf,608,256, P[22], 256,1, 256,256, 0,0, 0.f);
    lin(hA,608, xbuf,608,512, P[23], 16,3,  512,16,  0,0, 0.f);
    lin(hA,608, xbuf,608,560, P[24], 16,3,  560,16,  0,0, 0.f);
    zero(agg, (size_t)N*2176*4);
    repack(P[35], tpwT, 128, 128, 28000);
    tp_kernel<<<dim3(70, ET, 25), 32, 0, stream>>>(wq16, tpwT, D2, xbuf, agg, esh, elist, offs, cnt, esrc, edst);
    lin(agg,2176, hB,640,0,   P[25], 256,1, 0,256,    2096,16, 0.1f);
    lin(agg,2176, hB,640,256, P[26], 256,1, 1024,256, 2160,16, 0.1f);
    lin(agg,2176, hB,640,512, P[27], 32,1,  0,256,    2096,16, 0.1f);
    lin(agg,2176, hB,640,544, P[28], 16,3,  256,256,  2048,16, 0.1f);
    lin(agg,2176, hB,640,592, P[29], 16,3,  1280,256, 2112,16, 0.1f);
    gate_kernel<<<(N*608 + 255)/256, 256, 0, stream>>>(hB, 640, hA, 608, 256, 256, N);

    // ================= layer 3 (final, out N x 2) =================
    zero(hB, (size_t)N*2*4);
    lin(hA,608, hB,2,0, P[42], 1,1, 0,256,   0,0, 0.f);
    lin(hA,608, hB,2,1, P[43], 1,1, 256,256, 0,0, 0.f);
    lin(hA,608, xbuf,608,0,   P[36], 256,1, 0,256,   0,0, 0.f);
    lin(hA,608, xbuf,608,256, P[37], 256,1, 256,256, 0,0, 0.f);
    lin(hA,608, xbuf,608,512, P[38], 16,3,  512,16,  0,0, 0.f);
    lin(hA,608, xbuf,608,560, P[39], 16,3,  560,16,  0,0, 0.f);
    zero(agg, (size_t)N*544*4);
    repack(P[44], tpwT, 128, 128, 13600);
    tp_kernel<<<dim3(34, ET, 25), 32, 0, stream>>>(wq16, tpwT, D3, xbuf, agg, esh, elist, offs, cnt, esrc, edst);
    lin(agg,544, hB,2,0, P[40], 1,1, 0,256,   512,16, 0.1f);
    lin(agg,544, hB,2,1, P[41], 1,1, 256,256, 528,16, 0.1f);

    readout_kernel<<<(N + 255)/256, 256, 0, stream>>>(hB, batch, (float*)d_out, N);
}
